// SimpleLSTM_9895604650238
// MI455X (gfx1250) — matrix operand order repaired and checked by reading, not compiled, not run on hardware
//
#include <hip/hip_runtime.h>
#include <hip/hip_bf16.h>

// ---------------------------------------------------------------------------
// 3-layer LSTM (B=8192, T=336, F=8, H=20) + dense head, for gfx1250 (CDNA5).
//
// One wave32 owns 16 batch rows for a whole layer; h (fp16 B-fragment) and
// c (fp32) live in registers across all 336 steps. Per step:
//   gates[80 x 16b] = Wih_T(80x32,f16) x xfrag(32x16,f16)      (5 WMMAs)
//                   + Whh_T(80x32,f16) x hfrag(32x16,f16)      (5 WMMAs)
// Gate rows reordered as 4*unit+{i,f,g,o} so each lane's f32 D fragment holds
// all four gates of the same hidden unit -> lane-local activations.
// Activations use v_rcp_f32 / v_exp_f32 directly (no IEEE div sequences).
// ---------------------------------------------------------------------------

typedef __attribute__((ext_vector_type(16))) _Float16 v16h;
typedef __attribute__((ext_vector_type(8)))  float    v8f;

#define BATCH 8192
#define TLEN  336
#define FDIM  8
#define HDIM  20

// ws layout (bytes)
#define HSEQ_HALFS   ((size_t)BATCH * TLEN * 32)          // [t][b][32 units f16]
#define HSEQ_BYTES   (HSEQ_HALFS * 2)                     // 176,160,768
#define WFRAG_HALFS  ((size_t)3 * 2 * 5 * 32 * 16)
#define WFRAG_BYTES  (WFRAG_HALFS * 2)
#define BIAS_FLOATS  ((size_t)3 * 5 * 16)
#define WS_NEEDED    (HSEQ_BYTES + WFRAG_BYTES + BIAS_FLOATS * 4)

#define LOG2E 1.4426950408889634f

// single-instruction transcendental paths (v_rcp_f32 / v_exp_f32)
__device__ __forceinline__ float fast_rcp(float x)  { return __builtin_amdgcn_rcpf(x); }
__device__ __forceinline__ float fast_exp2(float x) { return __builtin_amdgcn_exp2f(x); }

// sigmoid(x) = 1 / (1 + 2^(-x*log2e))
__device__ __forceinline__ float fsigmoid(float x) {
    return fast_rcp(1.0f + fast_exp2(x * -LOG2E));
}
// tanh(x) = 1 - 2 / (2^(2x*log2e) + 1)   (saturates cleanly at +/-1)
__device__ __forceinline__ float ftanh_fast(float x) {
    return 1.0f - 2.0f * fast_rcp(fast_exp2(x * (2.0f * LOG2E)) + 1.0f);
}

// ---------------------------------------------------------------------------
// Pre-swizzle weights into per-lane A-fragment register images.
// A-fragment 16x32 f16, contiguous k order (B uses the same): lane group g=L/16, half q in 0..15:
//   K = 16*g + q,  M = L%16.
// Gate-row reorder: tile m row j -> unit = 4m + j/4, gate = j%4,
// original row = gate*20 + unit  (PyTorch i,f,g,o blocks of 20).
// ---------------------------------------------------------------------------
__global__ void lstm_prep_kernel(
    const float* Wih0, const float* Whh0, const float* bih0, const float* bhh0,
    const float* Wih1, const float* Whh1, const float* bih1, const float* bhh1,
    const float* Wih2, const float* Whh2, const float* bih2, const float* bhh2,
    _Float16* __restrict__ wfrag, float* __restrict__ biasbuf)
{
    const float* Wihs[3] = {Wih0, Wih1, Wih2};
    const float* Whhs[3] = {Whh0, Whh1, Whh2};
    const float* bihs[3] = {bih0, bih1, bih2};
    const float* bhhs[3] = {bhh0, bhh1, bhh2};

    int blk = blockIdx.x;            // 0..29
    int l = blk / 10;
    int rem = blk % 10;
    int w = rem / 5;                 // 0 = W_ih, 1 = W_hh
    int m = rem % 5;                 // gate tile (units 4m..4m+3)
    int lane = threadIdx.x;          // 0..31
    int g = lane >> 4;
    int j = lane & 15;

    const float* W = (w == 0) ? Wihs[l] : Whhs[l];
    int insz = (w == 0) ? ((l == 0) ? FDIM : HDIM) : HDIM;

    int unit = 4 * m + (j >> 2);
    int gate = j & 3;
    int row = gate * HDIM + unit;

    _Float16* outp = wfrag + (size_t)(((l * 2 + w) * 5 + m) * 512) + (size_t)lane * 16;
    #pragma unroll
    for (int q = 0; q < 16; ++q) {
        int K = 16 * g + q;
        float v = (K < insz) ? W[row * insz + K] : 0.0f;
        outp[q] = (_Float16)v;
    }

    if (w == 0 && lane < 16) {
        biasbuf[(l * 5 + m) * 16 + j] = bihs[l][row] + bhhs[l][row];
    }
}

// ---------------------------------------------------------------------------
// One LSTM layer. 128 threads = 4 waves per block; each wave owns 16 batches.
// grid = 8192 / (16*4) = 128 blocks.
// ---------------------------------------------------------------------------
__global__ __launch_bounds__(128) void lstm_layer_kernel(
    int layer,
    const float* __restrict__ x,                 // [B,T,F] f32 (layer 0 only)
    _Float16* __restrict__ hseq,                 // [T][B][32] f16 (in-place across layers)
    const _Float16* __restrict__ wfrag,
    const float* __restrict__ biasbuf)
{
    const int lane = threadIdx.x & 31;
    const int wid  = blockIdx.x * (blockDim.x >> 5) + (threadIdx.x >> 5);
    const int g    = lane >> 4;                  // lane group 0/1
    const int b    = wid * 16 + (lane & 15);     // batch row this lane's column maps to

    // --- weight A-fragments: resident in VGPRs for all 336 steps ---
    v16h a_ih[5], a_hh[5];
    v8f  biasf[5];
    const _Float16* wl = wfrag + (size_t)layer * 10 * 512;
    #pragma unroll
    for (int m = 0; m < 5; ++m) {
        a_ih[m] = *(const v16h*)(wl + (size_t)m * 512 + (size_t)lane * 16);
        a_hh[m] = *(const v16h*)(wl + (size_t)(5 + m) * 512 + (size_t)lane * 16);
        #pragma unroll
        for (int jj = 0; jj < 8; ++jj)
            biasf[m][jj] = biasbuf[(layer * 5 + m) * 16 + jj + 8 * g];
    }

    // per-lane state: 2 hidden units per tile (units 4m+2g, 4m+2g+1), batch b
    float c[10], hv[10];
    #pragma unroll
    for (int i = 0; i < 10; ++i) { c[i] = 0.0f; hv[i] = 0.0f; }
    v16h hfrag;
    #pragma unroll
    for (int q = 0; q < 16; ++q) hfrag[q] = (_Float16)0.0f;

    const float xmask = (g == 0) ? 1.0f : 0.0f;

    #pragma unroll 1
    for (int t = 0; t < TLEN; ++t) {
        // ---- input fragment (B-matrix 32x16: lanes 0-15 K=0..15, 16-31 K=16..31) ----
        v16h xfrag;
        if (layer == 0) {
            const float* xp = x + ((size_t)b * TLEN + t) * FDIM;
            #pragma unroll
            for (int q = 0; q < 8; ++q) xfrag[q] = (_Float16)(xp[q] * xmask);
            #pragma unroll
            for (int q = 8; q < 16; ++q) xfrag[q] = (_Float16)0.0f;
        } else {
            xfrag = *(const v16h*)(hseq + ((size_t)t * BATCH + b) * 32 + (size_t)g * 16);
        }

        // ---- 5 independent gate-tile chains: maximize WMMA ILP ----
        v8f acc[5];
        #pragma unroll
        for (int m = 0; m < 5; ++m)
            acc[m] = __builtin_amdgcn_wmma_f32_16x16x32_f16(
                false, a_ih[m], false, xfrag, (short)0, biasf[m], false, false);
        #pragma unroll
        for (int m = 0; m < 5; ++m)
            acc[m] = __builtin_amdgcn_wmma_f32_16x16x32_f16(
                false, a_hh[m], false, hfrag, (short)0, acc[m], false, false);

        // ---- lane-local activations: acc[m][4u+0..3] = i,f,g,o of unit 4m+2g+u ----
        #pragma unroll
        for (int m = 0; m < 5; ++m) {
            #pragma unroll
            for (int u = 0; u < 2; ++u) {
                float gi = fsigmoid(acc[m][4 * u + 0]);
                float gf = fsigmoid(acc[m][4 * u + 1]);
                float gg = ftanh_fast(acc[m][4 * u + 2]);
                float go = fsigmoid(acc[m][4 * u + 3]);
                float cv = gf * c[2 * m + u] + gi * gg;
                c[2 * m + u] = cv;
                hv[2 * m + u] = go * ftanh_fast(cv);
            }
        }

        // ---- rebuild h as fp16 B-fragment: exchange units with partner lane ----
        float pv[10];
        #pragma unroll
        for (int i = 0; i < 10; ++i) pv[i] = __shfl_xor(hv[i], 16, 32);

        v16h nh;
        #pragma unroll
        for (int q = 0; q < 16; ++q) nh[q] = (_Float16)0.0f;
        if (g == 0) {
            #pragma unroll
            for (int m = 0; m < 4; ++m) {
                nh[4 * m + 0] = (_Float16)hv[2 * m + 0];   // unit 4m   (self)
                nh[4 * m + 1] = (_Float16)hv[2 * m + 1];   // unit 4m+1 (self)
                nh[4 * m + 2] = (_Float16)pv[2 * m + 0];   // unit 4m+2 (partner)
                nh[4 * m + 3] = (_Float16)pv[2 * m + 1];   // unit 4m+3 (partner)
            }
        } else {
            nh[0] = (_Float16)pv[8];                       // unit 16 (partner)
            nh[1] = (_Float16)pv[9];                       // unit 17 (partner)
            nh[2] = (_Float16)hv[8];                       // unit 18 (self)
            nh[3] = (_Float16)hv[9];                       // unit 19 (self)
        }
        hfrag = nh;

        // store in the exact fragment layout the next layer (and head) loads
        *(v16h*)(hseq + ((size_t)t * BATCH + b) * 32 + (size_t)g * 16) = nh;
    }
}

// ---------------------------------------------------------------------------
// Head: relu(h_last @ fc1_w.T + fc1_b) @ fc2_w.T + fc2_b, one thread / batch.
// ---------------------------------------------------------------------------
__global__ void lstm_head_kernel(
    const _Float16* __restrict__ hseq,
    const float* __restrict__ fc1_w, const float* __restrict__ fc1_b,
    const float* __restrict__ fc2_w, const float* __restrict__ fc2_b,
    float* __restrict__ out)
{
    int b = blockIdx.x * blockDim.x + threadIdx.x;
    if (b >= BATCH) return;
    const _Float16* hp = hseq + ((size_t)(TLEN - 1) * BATCH + b) * 32;
    float h[HDIM];
    #pragma unroll
    for (int u = 0; u < HDIM; ++u) h[u] = (float)hp[u];

    float acc2 = fc2_b[0];
    #pragma unroll 4
    for (int d = 0; d < HDIM; ++d) {
        float s = fc1_b[d];
        #pragma unroll
        for (int u = 0; u < HDIM; ++u) s += fc1_w[d * HDIM + u] * h[u];
        s = fmaxf(s, 0.0f);
        acc2 += fc2_w[d] * s;
    }
    out[b] = acc2;
}

// ---------------------------------------------------------------------------
extern "C" void kernel_launch(void* const* d_in, const int* in_sizes, int n_in,
                              void* d_out, int out_size, void* d_ws, size_t ws_size,
                              hipStream_t stream) {
    (void)in_sizes; (void)n_in; (void)out_size;
    if (ws_size < WS_NEEDED) return;   // deterministic no-op if scratch too small

    const float* x     = (const float*)d_in[0];
    const float* Wih0  = (const float*)d_in[1];
    const float* Whh0  = (const float*)d_in[2];
    const float* bih0  = (const float*)d_in[3];
    const float* bhh0  = (const float*)d_in[4];
    const float* Wih1  = (const float*)d_in[5];
    const float* Whh1  = (const float*)d_in[6];
    const float* bih1  = (const float*)d_in[7];
    const float* bhh1  = (const float*)d_in[8];
    const float* Wih2  = (const float*)d_in[9];
    const float* Whh2  = (const float*)d_in[10];
    const float* bih2  = (const float*)d_in[11];
    const float* bhh2  = (const float*)d_in[12];
    const float* fc1_w = (const float*)d_in[13];
    const float* fc1_b = (const float*)d_in[14];
    const float* fc2_w = (const float*)d_in[15];
    const float* fc2_b = (const float*)d_in[16];

    char* ws = (char*)d_ws;
    _Float16* hseq   = (_Float16*)ws;
    _Float16* wfrag  = (_Float16*)(ws + HSEQ_BYTES);
    float*    biasbf = (float*)(ws + HSEQ_BYTES + WFRAG_BYTES);

    lstm_prep_kernel<<<30, 32, 0, stream>>>(
        Wih0, Whh0, bih0, bhh0, Wih1, Whh1, bih1, bhh1, Wih2, Whh2, bih2, bhh2,
        wfrag, biasbf);

    // 128 blocks x 128 threads = 512 waves, each owns 16 batches.
    for (int l = 0; l < 3; ++l) {
        lstm_layer_kernel<<<128, 128, 0, stream>>>(l, x, hseq, wfrag, biasbf);
    }

    lstm_head_kernel<<<BATCH / 256, 256, 0, stream>>>(
        hseq, fc1_w, fc1_b, fc2_w, fc2_b, (float*)d_out);
}
